// MaskedDiffusionResidual_53120155517396
// MI455X (gfx1250) — matrix operand order repaired and checked by reading, not compiled, not run on hardware
//
#include <hip/hip_runtime.h>
#include <hip/hip_bf16.h>
#include <math.h>

// ---------------------------------------------------------------------------
// MaskedDiffusionResidual for MI455X (gfx1250): all GEMMs + attention matmuls
// run on v_wmma_f32_16x16x32_f16 (f16 in, f32 accumulate). Compute-bound
// (~720 GFLOP total); weights (~7MB f16) + activations (~100MB) live in L2.
// GEMM register-blocks 32x32 per wave: 4 back-to-back WMMAs per K-chunk with
// 2x A/B fragment reuse; tiles staged with global_load_async_to_lds_b128
// (ASYNCcnt-tracked) when the toolchain provides the builtin.
// ---------------------------------------------------------------------------

#define T_TOK   8192          // B*S = 8*1024
#define S_LEN   1024
#define NB      8
#define DIMV    256
#define NHEADS  8
#define HDIM    32
#define FFNV    1024
#define NCODE   1024
#define MASKID  1024
#define DEPTHV  4
#define NSTEPS  8

// CDNA5 async global->LDS copy (ASYNCcnt-tracked), if this toolchain has it.
#if defined(__has_builtin)
#if __has_builtin(__builtin_amdgcn_global_load_async_to_lds_b128) && \
    __has_builtin(__builtin_amdgcn_s_wait_asynccnt)
#define HAVE_ASYNC_LDS 1
#endif
#endif

typedef __attribute__((ext_vector_type(16))) _Float16 v16h;
typedef __attribute__((ext_vector_type(8)))  float    v8f;
typedef int v4i __attribute__((vector_size(16)));
typedef v4i __attribute__((address_space(1)))* gptr_b128;
typedef v4i __attribute__((address_space(3)))* lptr_b128;

union F16x16 { v16h v; _Float16 h[16]; unsigned u[8]; };
union F32x8  { v8f  v; float f[8]; };

__device__ __forceinline__ v8f wmma_f16(v16h a, v16h b, v8f c) {
  // 8 args: (neg_a, A, neg_b, B, c_mod, C, reuse_a, reuse_b)
  return __builtin_amdgcn_wmma_f32_16x16x32_f16(false, a, false, b, (short)0, c,
                                                false, false);
}

__device__ __forceinline__ void stage16(const _Float16* g, _Float16* l) {
#ifdef HAVE_ASYNC_LDS
  __builtin_amdgcn_global_load_async_to_lds_b128(
      (gptr_b128)(_Float16*)g, (lptr_b128)l, 0, 0);
#else
  *(float4*)l = *(const float4*)g;
#endif
}

__device__ __forceinline__ void stage_wait() {
#ifdef HAVE_ASYNC_LDS
  __builtin_amdgcn_s_wait_asynccnt(0);
#endif
}

__device__ __forceinline__ float blockReduceSum(float v, float* sh) {
  int lane = threadIdx.x & 31, w = threadIdx.x >> 5;
#pragma unroll
  for (int m = 16; m >= 1; m >>= 1) v += __shfl_xor(v, m, 32);
  if (lane == 0) sh[w] = v;
  __syncthreads();
  if (threadIdx.x == 0) {
    float s = 0.f;
    int nw = (blockDim.x + 31) >> 5;
    for (int i = 0; i < nw; ++i) s += sh[i];
    sh[0] = s;
  }
  __syncthreads();
  float r = sh[0];
  __syncthreads();
  return r;
}

// -------------------------- f32 -> f16 convert -----------------------------
__global__ void k_convert(const float* __restrict__ src,
                          _Float16* __restrict__ dst, int n) {
  int i = blockIdx.x * 256 + threadIdx.x;
  if (i < n) dst[i] = (_Float16)src[i];
}

// -------------------------- idx init ---------------------------------------
__global__ void k_init_idx(const int* __restrict__ trans,
                           const unsigned char* __restrict__ keep,
                           int* __restrict__ idx) {
  int t = blockIdx.x * 256 + threadIdx.x;
  if (t < T_TOK) idx[t] = keep[t] ? trans[t] : MASKID;
}

// -------------------------- context: pool + 1x1 proj -----------------------
__global__ void __launch_bounds__(256)
k_ctx(const float* __restrict__ xmc, const float* __restrict__ ctxW,
      const float* __restrict__ ctxB, float* __restrict__ ctx) {
  __shared__ float partial[192];
  __shared__ float pool[3];
  int t = blockIdx.x;
  int b = t >> 10, s = t & 1023, hh = s >> 5, ww = s & 31;
  int tid = threadIdx.x;
  if (tid < 192) {
    int c = tid / 64, e = tid % 64, pr = e >> 3, pc = e & 7;
    partial[tid] =
        xmc[(((size_t)(b * 3 + c) * 256) + hh * 8 + pr) * 256 + ww * 8 + pc];
  }
  __syncthreads();
  if (tid < 3) {
    float sum = 0.f;
    for (int e = 0; e < 64; ++e) sum += partial[tid * 64 + e];
    pool[tid] = sum * (1.0f / 64.0f);
  }
  __syncthreads();
  float v = ctxB[tid];
#pragma unroll
  for (int c = 0; c < 3; ++c) v += pool[c] * ctxW[tid * 3 + c];
  ctx[(size_t)t * DIMV + tid] = v;
}

// -------------------------- embedding --------------------------------------
__global__ void __launch_bounds__(256)
k_embed(const int* __restrict__ idx, const float* __restrict__ tokEmb,
        const float* __restrict__ posEmb, const float* __restrict__ ctx,
        float* __restrict__ h32, _Float16* __restrict__ h16) {
  int t = blockIdx.x, d = threadIdx.x;
  int id = idx[t];
  float v = tokEmb[(size_t)id * DIMV + d] + ctx[(size_t)t * DIMV + d] +
            posEmb[(size_t)(t & 1023) * DIMV + d];
  h32[(size_t)t * DIMV + d] = v;
  h16[(size_t)t * DIMV + d] = (_Float16)v;
}

// -------------------------- WMMA GEMM: C = act(A @ W^T + bias) -------------
// A: [M,K] f16 row-major. W: [N,K] f16 row-major (== B operand directly).
// Block tile 128(M) x 64(N); 8 waves in 4x2 grid; each wave register-blocks
// a 32x32 C tile = 2x2 WMMA tiles (4 v_wmma per K-chunk, frags reused 2x).
__global__ void __launch_bounds__(256)
k_gemm(const _Float16* __restrict__ A, const _Float16* __restrict__ W,
       const float* __restrict__ bias, float* __restrict__ outF32,
       _Float16* __restrict__ outF16, int M, int N, int K, int gelu) {
  __shared__ alignas(16) _Float16 As[128][32];
  __shared__ alignas(16) _Float16 Ws[64][32];
  const int tid = threadIdx.x;
  const int lane = tid & 31, wave = tid >> 5;
  const int wm = wave >> 1, wn = wave & 1;
  const int mBase = blockIdx.y * 128, nBase = blockIdx.x * 64;
  const int half = lane >> 4, l16 = lane & 15;

  F32x8 acc[2][2];
#pragma unroll
  for (int i = 0; i < 2; ++i)
#pragma unroll
    for (int j = 0; j < 2; ++j)
#pragma unroll
      for (int r = 0; r < 8; ++r) acc[i][j].f[r] = 0.f;

  const int sr = tid >> 2, sc = (tid & 3) * 8;  // 16B chunk per thread
  for (int k0 = 0; k0 < K; k0 += 32) {
    // stage A tile (128x32): two 16B chunks per thread
    stage16(A + (size_t)(mBase + sr) * K + k0 + sc, &As[sr][sc]);
    stage16(A + (size_t)(mBase + 64 + sr) * K + k0 + sc, &As[64 + sr][sc]);
    // stage W tile (64x32): one 16B chunk per thread
    stage16(W + (size_t)(nBase + sr) * K + k0 + sc, &Ws[sr][sc]);
    if (k0 + 32 < K) {  // lowers to global_prefetch on gfx1250
      __builtin_prefetch(A + (size_t)(mBase + sr) * K + k0 + 32 + sc, 0, 3);
      __builtin_prefetch(W + (size_t)(nBase + sr) * K + k0 + 32 + sc, 0, 3);
    }
    stage_wait();
    __syncthreads();

    F16x16 af[2], bf[2];
#pragma unroll
    for (int i = 0; i < 2; ++i) {
      int row = wm * 32 + i * 16 + l16, kh = half * 8;
#pragma unroll
      for (int j = 0; j < 4; ++j)
        af[i].u[j] = *(const unsigned*)&As[row][kh + 2 * j];
#pragma unroll
      for (int j = 0; j < 4; ++j)
        af[i].u[4 + j] = *(const unsigned*)&As[row][16 + kh + 2 * j];
    }
#pragma unroll
    for (int i = 0; i < 2; ++i) {
      int row = wn * 32 + i * 16 + l16, kb = half * 8;
#pragma unroll
      for (int j = 0; j < 4; ++j)
        bf[i].u[j] = *(const unsigned*)&Ws[row][kb + 2 * j];
#pragma unroll
      for (int j = 0; j < 4; ++j)
        bf[i].u[4 + j] = *(const unsigned*)&Ws[row][16 + kb + 2 * j];
    }
    // 4 back-to-back WMMAs, each fragment reused twice
    acc[0][0].v = wmma_f16(af[0].v, bf[0].v, acc[0][0].v);
    acc[0][1].v = wmma_f16(af[0].v, bf[1].v, acc[0][1].v);
    acc[1][0].v = wmma_f16(af[1].v, bf[0].v, acc[1][0].v);
    acc[1][1].v = wmma_f16(af[1].v, bf[1].v, acc[1][1].v);
    __syncthreads();
  }

#pragma unroll
  for (int j = 0; j < 2; ++j) {
    int nG = nBase + wn * 32 + j * 16 + l16;
    float bv = bias ? bias[nG] : 0.f;
#pragma unroll
    for (int i = 0; i < 2; ++i) {
#pragma unroll
      for (int r = 0; r < 8; ++r) {
        int m = mBase + wm * 32 + i * 16 + half * 8 + r;
        float v = acc[i][j].f[r] + bv;
        if (gelu) v = 0.5f * v * (1.0f + erff(v * 0.70710678118654752f));
        if (outF32) outF32[(size_t)m * N + nG] = v;
        if (outF16) outF16[(size_t)m * N + nG] = (_Float16)v;
      }
    }
  }
}

// -------------------------- attention (one wave / 16-query tile) -----------
// qkv: [T, 768] f16 (q|k|v each 256 cols, head h at col h*32).
__global__ void __launch_bounds__(32)
k_attn(const _Float16* __restrict__ qkv, _Float16* __restrict__ attnOut) {
  __shared__ alignas(16) _Float16 P[16][S_LEN];  // 32KB score/prob tile
  const int lane = threadIdx.x;
  const int half = lane >> 4, l16 = lane & 15;
  const int q0 = blockIdx.x * 16;
  const int hd0 = blockIdx.y * HDIM;
  const int b = blockIdx.z;

  // Q fragment (A operand, 16x32)
  F16x16 qa;
  {
    const _Float16* qp = qkv + (size_t)(b * S_LEN + q0 + l16) * 768 + hd0;
    int kh = half * 8;
#pragma unroll
    for (int j = 0; j < 4; ++j) qa.u[j] = *(const unsigned*)(qp + kh + 2 * j);
#pragma unroll
    for (int j = 0; j < 4; ++j)
      qa.u[4 + j] = *(const unsigned*)(qp + 16 + kh + 2 * j);
  }

  // scores = Q @ K^T * 1/sqrt(32)
  v8f zero = {0.f, 0.f, 0.f, 0.f, 0.f, 0.f, 0.f, 0.f};
  for (int kk = 0; kk < S_LEN; kk += 16) {
    F16x16 kb;
    const _Float16* kp =
        qkv + (size_t)(b * S_LEN + kk + l16) * 768 + 256 + hd0 + half * 8;
#pragma unroll
    for (int j = 0; j < 4; ++j) kb.u[j] = *(const unsigned*)(kp + 2 * j);
#pragma unroll
    for (int j = 0; j < 4; ++j)
      kb.u[4 + j] = *(const unsigned*)(kp + 16 + 2 * j);
    F32x8 s;
    s.v = wmma_f16(qa.v, kb.v, zero);
#pragma unroll
    for (int r = 0; r < 8; ++r)
      P[r + half * 8][kk + l16] = (_Float16)(s.f[r] * 0.17677669529663689f);
  }
  __syncthreads();

  // softmax rows (wave32 shuffle reductions)
  for (int m = 0; m < 16; ++m) {
    float mx = -1e30f;
    for (int i = lane; i < S_LEN; i += 32) mx = fmaxf(mx, (float)P[m][i]);
#pragma unroll
    for (int sh = 16; sh >= 1; sh >>= 1) mx = fmaxf(mx, __shfl_xor(mx, sh, 32));
    float sum = 0.f;
    for (int i = lane; i < S_LEN; i += 32) sum += expf((float)P[m][i] - mx);
#pragma unroll
    for (int sh = 16; sh >= 1; sh >>= 1) sum += __shfl_xor(sum, sh, 32);
    float inv = 1.0f / sum;
    for (int i = lane; i < S_LEN; i += 32)
      P[m][i] = (_Float16)(expf((float)P[m][i] - mx) * inv);
  }
  __syncthreads();

  // O = P @ V  (two 16-wide N halves of head_dim 32)
  F32x8 accs[2];
#pragma unroll
  for (int r = 0; r < 8; ++r) { accs[0].f[r] = 0.f; accs[1].f[r] = 0.f; }
  for (int kk = 0; kk < S_LEN; kk += 32) {
    F16x16 pa;
    {
      int kh = half * 8;
#pragma unroll
      for (int j = 0; j < 4; ++j)
        pa.u[j] = *(const unsigned*)&P[l16][kk + kh + 2 * j];
#pragma unroll
      for (int j = 0; j < 4; ++j)
        pa.u[4 + j] = *(const unsigned*)&P[l16][kk + 16 + kh + 2 * j];
    }
#pragma unroll
    for (int nh = 0; nh < 2; ++nh) {
      F16x16 vb;
      int hdc = hd0 + nh * 16 + l16;
      int keyb = b * S_LEN + kk + half * 8;
#pragma unroll
      for (int j = 0; j < 8; ++j) {
        int ko = 2 * j + (j & 4) * 2;
        vb.h[2 * j] = qkv[(size_t)(keyb + ko) * 768 + 512 + hdc];
        vb.h[2 * j + 1] = qkv[(size_t)(keyb + ko + 1) * 768 + 512 + hdc];
      }
      accs[nh].v = wmma_f16(pa.v, vb.v, accs[nh].v);
    }
  }
#pragma unroll
  for (int r = 0; r < 8; ++r) {
    size_t t = (size_t)(b * S_LEN + q0 + half * 8 + r) * DIMV;
    attnOut[t + hd0 + l16] = (_Float16)accs[0].f[r];
    attnOut[t + hd0 + 16 + l16] = (_Float16)accs[1].f[r];
  }
}

// -------------------------- residual + LayerNorm ---------------------------
__global__ void __launch_bounds__(256)
k_add_ln(float* __restrict__ h32, const float* __restrict__ res,
         const float* __restrict__ g, const float* __restrict__ be,
         _Float16* __restrict__ h16) {
  __shared__ float red[8];
  int t = blockIdx.x, d = threadIdx.x;
  float v = h32[(size_t)t * DIMV + d] + res[(size_t)t * DIMV + d];
  float mu = blockReduceSum(v, red) * (1.0f / DIMV);
  float dv = v - mu;
  float var = blockReduceSum(dv * dv, red) * (1.0f / DIMV);
  float y = dv * rsqrtf(var + 1e-5f) * g[d] + be[d];
  h32[(size_t)t * DIMV + d] = y;
  h16[(size_t)t * DIMV + d] = (_Float16)y;
}

// -------------------------- sampling: argmax + confidence ------------------
__global__ void __launch_bounds__(256)
k_sample(const float* __restrict__ logits, const int* __restrict__ idx,
         int* __restrict__ sampled, float* __restrict__ confM) {
  __shared__ float redv[8];
  __shared__ int redi[8];
  __shared__ float sh[8];
  __shared__ float mxs;
  __shared__ int args;
  int t = blockIdx.x, tid = threadIdx.x;
  const float* lg = logits + (size_t)t * NCODE;
  float mv = -1e30f;
  int mi = 0;
  for (int i = tid; i < NCODE; i += 256) {
    float v = lg[i];
    if (v > mv || (v == mv && i < mi)) { mv = v; mi = i; }
  }
#pragma unroll
  for (int m = 16; m >= 1; m >>= 1) {
    float ov = __shfl_xor(mv, m, 32);
    int oi = __shfl_xor(mi, m, 32);
    if (ov > mv || (ov == mv && oi < mi)) { mv = ov; mi = oi; }
  }
  int lane = tid & 31, w = tid >> 5;
  if (lane == 0) { redv[w] = mv; redi[w] = mi; }
  __syncthreads();
  if (tid == 0) {
    float bm = redv[0]; int bi = redi[0];
    for (int i = 1; i < 8; ++i)
      if (redv[i] > bm || (redv[i] == bm && redi[i] < bi)) {
        bm = redv[i]; bi = redi[i];
      }
    mxs = bm; args = bi;
  }
  __syncthreads();
  float mx = mxs;
  float sum = 0.f;
  for (int i = tid; i < NCODE; i += 256) sum += expf(lg[i] - mx);
  sum = blockReduceSum(sum, sh);
  if (tid == 0) {
    sampled[t] = args;
    confM[t] = (idx[t] == MASKID) ? (1.0f / sum) : -1.0f;  // conf of argmax
  }
}

// -------------------------- exact k-th largest (radix select) --------------
__global__ void __launch_bounds__(256)
k_topk(const float* __restrict__ confM, float* __restrict__ thresh, int k) {
  __shared__ unsigned keys[S_LEN];
  __shared__ int red[8];
  int b = blockIdx.x, tid = threadIdx.x;
  for (int i = tid; i < S_LEN; i += 256) {
    unsigned u = __float_as_uint(confM[b * S_LEN + i]);
    keys[i] = (u & 0x80000000u) ? ~u : (u | 0x80000000u);
  }
  __syncthreads();
  unsigned prefix = 0;
  for (int bit = 31; bit >= 0; --bit) {
    unsigned cand = prefix | (1u << bit);
    int c = 0;
    for (int i = tid; i < S_LEN; i += 256) c += (keys[i] >= cand) ? 1 : 0;
#pragma unroll
    for (int m = 16; m >= 1; m >>= 1) c += __shfl_xor(c, m, 32);
    int lane = tid & 31, w = tid >> 5;
    if (lane == 0) red[w] = c;
    __syncthreads();
    if (tid == 0) {
      int tot = 0;
      for (int i = 0; i < 8; ++i) tot += red[i];
      red[0] = tot;
    }
    __syncthreads();
    int tot = red[0];
    __syncthreads();
    if (tot >= k) prefix = cand;
  }
  if (tid == 0) {
    unsigned u = (prefix & 0x80000000u) ? (prefix ^ 0x80000000u) : ~prefix;
    thresh[b] = __uint_as_float(u);
  }
}

// -------------------------- fill masked tokens -----------------------------
__global__ void k_fill(int* __restrict__ idx, const int* __restrict__ sampled,
                       const float* __restrict__ confM,
                       const float* __restrict__ thresh) {
  int t = blockIdx.x * 256 + threadIdx.x;
  if (t >= T_TOK) return;
  if (idx[t] == MASKID && confM[t] >= thresh[t >> 10]) idx[t] = sampled[t];
}

// -------------------------- decoder (codebook + 8x8 deconv) ----------------
__global__ void __launch_bounds__(192)
k_decode(const int* __restrict__ idx, const float* __restrict__ codebook,
         const float* __restrict__ decW, const float* __restrict__ decB,
         float* __restrict__ out) {
  __shared__ float z[64];
  int t = blockIdx.x, tid = threadIdx.x;
  int b = t >> 10, s = t & 1023, hh = s >> 5, ww = s & 31;
  if (tid < 64) z[tid] = codebook[(size_t)idx[t] * 64 + tid];
  __syncthreads();
  int c = tid / 64, kl = tid % 64, kk = kl >> 3, ll = kl & 7;
  float acc = decB[c];
#pragma unroll 8
  for (int d = 0; d < 64; ++d) acc += z[d] * decW[d * 192 + tid];
  out[(((size_t)(b * 3 + c) * 256) + hh * 8 + kk) * 256 + ww * 8 + ll] = acc;
}

// ===========================================================================
extern "C" void kernel_launch(void* const* d_in, const int* in_sizes, int n_in,
                              void* d_out, int out_size, void* d_ws,
                              size_t ws_size, hipStream_t stream) {
  (void)in_sizes; (void)n_in; (void)out_size; (void)ws_size;
  const int* transmitted       = (const int*)d_in[0];
  const unsigned char* keep    = (const unsigned char*)d_in[1];  // bool array
  const float* xmc             = (const float*)d_in[2];
  const float* tokEmb          = (const float*)d_in[3];
  const float* posEmb          = (const float*)d_in[4];
  const float* ctxW            = (const float*)d_in[5];
  const float* ctxB            = (const float*)d_in[6];
  const float* qkvW            = (const float*)d_in[7];
  const float* qkvB            = (const float*)d_in[8];
  const float* outW            = (const float*)d_in[9];
  const float* outB            = (const float*)d_in[10];
  const float* ln1g            = (const float*)d_in[11];
  const float* ln1b            = (const float*)d_in[12];
  const float* ff1W            = (const float*)d_in[13];
  const float* ff1B            = (const float*)d_in[14];
  const float* ff2W            = (const float*)d_in[15];
  const float* ff2B            = (const float*)d_in[16];
  const float* ln2g            = (const float*)d_in[17];
  const float* ln2b            = (const float*)d_in[18];
  const float* headW           = (const float*)d_in[19];
  const float* headB           = (const float*)d_in[20];
  const float* codebook        = (const float*)d_in[21];
  const float* decW            = (const float*)d_in[22];
  const float* decB            = (const float*)d_in[23];
  float* out = (float*)d_out;

  // ---- workspace carve (256B aligned bump allocator) ----
  char* base = (char*)d_ws;
  size_t off = 0;
  auto carve = [&](size_t bytes) -> void* {
    void* p = base + off;
    off += (bytes + 255) & ~(size_t)255;
    return p;
  };
  _Float16* wqkv16  = (_Float16*)carve((size_t)DEPTHV * 768 * 256 * 2);
  _Float16* wout16  = (_Float16*)carve((size_t)DEPTHV * 256 * 256 * 2);
  _Float16* wff1_16 = (_Float16*)carve((size_t)DEPTHV * 1024 * 256 * 2);
  _Float16* wff2_16 = (_Float16*)carve((size_t)DEPTHV * 256 * 1024 * 2);
  _Float16* whead16 = (_Float16*)carve((size_t)1024 * 256 * 2);
  float*    ctx32   = (float*)carve((size_t)T_TOK * DIMV * 4);
  float*    h32     = (float*)carve((size_t)T_TOK * DIMV * 4);
  _Float16* h16     = (_Float16*)carve((size_t)T_TOK * DIMV * 2);
  _Float16* qkv16   = (_Float16*)carve((size_t)T_TOK * 768 * 2);
  _Float16* attn16  = (_Float16*)carve((size_t)T_TOK * DIMV * 2);
  float*    res32   = (float*)carve((size_t)T_TOK * DIMV * 4);
  _Float16* ff16buf = (_Float16*)carve((size_t)T_TOK * FFNV * 2);
  float*    logits  = (float*)carve((size_t)T_TOK * NCODE * 4);
  int*      idxBuf  = (int*)carve((size_t)T_TOK * 4);
  int*      sampled = (int*)carve((size_t)T_TOK * 4);
  float*    confM   = (float*)carve((size_t)T_TOK * 4);
  float*    thresh  = (float*)carve(NB * 4);

  // ---- one-time setup ----
  auto cvt = [&](const float* s, _Float16* d, int n) {
    k_convert<<<(n + 255) / 256, 256, 0, stream>>>(s, d, n);
  };
  cvt(qkvW, wqkv16, DEPTHV * 768 * 256);
  cvt(outW, wout16, DEPTHV * 256 * 256);
  cvt(ff1W, wff1_16, DEPTHV * 1024 * 256);
  cvt(ff2W, wff2_16, DEPTHV * 256 * 1024);
  cvt(headW, whead16, 1024 * 256);
  k_ctx<<<T_TOK, 256, 0, stream>>>(xmc, ctxW, ctxB, ctx32);
  k_init_idx<<<T_TOK / 256, 256, 0, stream>>>(transmitted, keep, idxBuf);

  // ---- masked-diffusion refinement ----
  for (int t = 0; t < NSTEPS; ++t) {
    k_embed<<<T_TOK, 256, 0, stream>>>(idxBuf, tokEmb, posEmb, ctx32, h32, h16);
    for (int l = 0; l < DEPTHV; ++l) {
      // QKV: [8192,256] x [768,256]^T
      k_gemm<<<dim3(768 / 64, T_TOK / 128), 256, 0, stream>>>(
          h16, wqkv16 + (size_t)l * 768 * 256, qkvB + l * 768, nullptr, qkv16,
          T_TOK, 768, 256, 0);
      // attention
      k_attn<<<dim3(S_LEN / 16, NHEADS, NB), 32, 0, stream>>>(qkv16, attn16);
      // output projection
      k_gemm<<<dim3(256 / 64, T_TOK / 128), 256, 0, stream>>>(
          attn16, wout16 + (size_t)l * 256 * 256, outB + l * 256, res32,
          nullptr, T_TOK, 256, 256, 0);
      k_add_ln<<<T_TOK, 256, 0, stream>>>(h32, res32, ln1g + l * 256,
                                          ln1b + l * 256, h16);
      // FFN
      k_gemm<<<dim3(1024 / 64, T_TOK / 128), 256, 0, stream>>>(
          h16, wff1_16 + (size_t)l * 1024 * 256, ff1B + l * 1024, nullptr,
          ff16buf, T_TOK, 1024, 256, 1);  // GELU
      k_gemm<<<dim3(256 / 64, T_TOK / 128), 256, 0, stream>>>(
          ff16buf, wff2_16 + (size_t)l * 256 * 1024, ff2B + l * 256, res32,
          nullptr, T_TOK, 256, 1024, 0);
      k_add_ln<<<T_TOK, 256, 0, stream>>>(h32, res32, ln2g + l * 256,
                                          ln2b + l * 256, h16);
    }
    // head logits
    k_gemm<<<dim3(NCODE / 64, T_TOK / 128), 256, 0, stream>>>(
        h16, whead16, headB, logits, nullptr, T_TOK, NCODE, 256, 0);
    k_sample<<<T_TOK, 256, 0, stream>>>(logits, idxBuf, sampled, confM);
    // schedule k = min(int(1024 * cos(pi/2*(1-(t+1)/8))), 1024), >= 1
    double frac = cos(3.14159265358979323846 * 0.5 *
                      (1.0 - (double)(t + 1) / (double)NSTEPS));
    int kk = (int)(1024.0 * frac);
    if (kk > 1024) kk = 1024;
    if (kk < 1) kk = 1;
    k_topk<<<NB, 256, 0, stream>>>(confM, thresh, kk);
    k_fill<<<T_TOK / 256, 256, 0, stream>>>(idxBuf, sampled, confM, thresh);
  }

  // ---- decode to [8,3,256,256] ----
  k_decode<<<T_TOK, 192, 0, stream>>>(idxBuf, codebook, decW, decB, out);
}
